// GeometryAwareAnchors_33397665693865
// MI455X (gfx1250) — matrix operand order repaired and checked by reading, not compiled, not run on hardware
//
#include <hip/hip_runtime.h>

typedef __attribute__((ext_vector_type(16))) _Float16 v16h;
typedef __attribute__((ext_vector_type(8)))  _Float16 v8h;
typedef __attribute__((ext_vector_type(4)))  _Float16 v4h;
typedef __attribute__((ext_vector_type(8)))  float    v8f;
typedef __attribute__((ext_vector_type(4)))  float    v4f;

#define BATCH_N   8192
#define EMBED_D   1024
#define NUM_CLS   64
#define NUM_ANC   32
#define CA_N      2048   // NUM_CLS * NUM_ANC
#define M_TILE    64     // rows per workgroup

#define LOG2E     1.4426950408889634f

// ---------------------------------------------------------------------------
// Kernel 1: f32 -> f16 conversion + per-row sum of squares (one wave per row)
// ---------------------------------------------------------------------------
__global__ __launch_bounds__(256) void cvt_rowsq_kernel(const float* __restrict__ src,
                                                        _Float16* __restrict__ dst,
                                                        float* __restrict__ rowsq,
                                                        int nrows)
{
    int gt   = blockIdx.x * blockDim.x + threadIdx.x;
    int row  = gt >> 5;          // wave32: one wave per row
    int lane = gt & 31;
    if (row >= nrows) return;

    const float* s = src + (size_t)row * EMBED_D;
    _Float16*    d = dst + (size_t)row * EMBED_D;

    float acc = 0.0f;
    #pragma unroll
    for (int i = lane * 4; i < EMBED_D; i += 128) {
        v4f v = *(const v4f*)(s + i);
        acc += v.x * v.x + v.y * v.y + v.z * v.z + v.w * v.w;
        v4h h;
        h.x = (_Float16)v.x; h.y = (_Float16)v.y;
        h.z = (_Float16)v.z; h.w = (_Float16)v.w;
        *(v4h*)(d + i) = h;
    }
    #pragma unroll
    for (int o = 16; o > 0; o >>= 1) acc += __shfl_xor(acc, o, 32);
    if (lane == 0) rowsq[row] = acc;
}

// ---------------------------------------------------------------------------
// Kernel 2: fused WMMA GEMM (X @ A^T) + distance + double-exp + class-softmax
//   block = 256 threads (8 waves); each block does 64 rows x all 2048 cols.
//   wave (w&1)  -> which 32-row half (2 WMMA M-tiles)
//   wave (w>>1) -> which 512-column quarter of the 2048 anchors
//   Per K-step: 4 A-frag loads + 8 B-frag loads feed 8 WMMAs (B reused 2x).
// ---------------------------------------------------------------------------
__global__ __launch_bounds__(256) void fused_anchor_kernel(
    const _Float16* __restrict__ Xh, const _Float16* __restrict__ Ah,
    const float* __restrict__ xsq, const float* __restrict__ asq,
    const int* __restrict__ tptr, float* __restrict__ out)
{
    __shared__ float clsAcc[M_TILE][NUM_CLS];   // 16 KB
    __shared__ float denom[M_TILE];

    const int tid = threadIdx.x;
    for (int i = tid; i < M_TILE * NUM_CLS; i += 256) (&clsAcc[0][0])[i] = 0.0f;
    __syncthreads();

    const int wave = tid >> 5;
    const int lane = tid & 31;
    const int hi   = lane >> 4;     // which 16-lane half of the wave
    const int l15  = lane & 15;

    const int mloc0 = (wave & 1) * 32;               // 2 M-tiles: mloc0, mloc0+16
    const int m0    = blockIdx.x * M_TILE + mloc0;
    const int nbase = (wave >> 1) * (CA_N / 4);      // 512-column quarter

    const float invT = 1.0f / (float)tptr[0];

    // xs[i][r] = ||x||^2 for the row element r of M-tile i maps to in this lane
    float xs[2][8];
    #pragma unroll
    for (int i = 0; i < 2; ++i)
        #pragma unroll
        for (int r = 0; r < 8; ++r)
            xs[i][r] = xsq[m0 + i * 16 + r + (hi << 3)];

    // A fragment (16x32 f16, ISA layout): lane<16 reads K 0-7 & 16-23 of row
    // M=l15; lane>=16 reads K 8-15 & 24-31 of the same row.
    const _Float16* arow0 = Xh + (size_t)(m0 + l15) * EMBED_D + (hi << 3);
    const _Float16* arow1 = arow0 + (size_t)16 * EMBED_D;

    for (int nb = 0; nb < CA_N / 4; nb += 64) {
        const int n0 = nbase + nb;
        v8f acc[2][4] = {};     // 2 M-tiles x 4 N-tiles -> 32x64 block

        // B fragment (32x16): lane holds column N=l15 = anchor row (n0+l15),
        // lane<16 reads K 0-7 & 16-23; lane>=16 reads K 8-15 & 24-31.
        const _Float16* brow = Ah + (size_t)(n0 + l15) * EMBED_D + (hi << 3);

        // Prefetch next n-block's anchor rows into cache (global_prefetch_b8)
        if (nb + 64 < CA_N / 4) {
            #pragma unroll
            for (int j = 0; j < 4; ++j)
                __builtin_prefetch(brow + (size_t)(64 + j * 16) * EMBED_D, 0, 1);
        }

        #pragma unroll 2
        for (int k = 0; k < EMBED_D; k += 32) {
            v8h a0lo = *(const v8h*)(arow0 + k);
            v8h a0hi = *(const v8h*)(arow0 + k + 16);
            v16h af0 = __builtin_shufflevector(a0lo, a0hi,
                         0,1,2,3,4,5,6,7,8,9,10,11,12,13,14,15);
            v8h a1lo = *(const v8h*)(arow1 + k);
            v8h a1hi = *(const v8h*)(arow1 + k + 16);
            v16h af1 = __builtin_shufflevector(a1lo, a1hi,
                         0,1,2,3,4,5,6,7,8,9,10,11,12,13,14,15);
            #pragma unroll
            for (int j = 0; j < 4; ++j) {
                const _Float16* bp = brow + (size_t)(j * 16) * EMBED_D + k;
                v8h blo = *(const v8h*)(bp);
                v8h bhi = *(const v8h*)(bp + 16);
                v16h bf = __builtin_shufflevector(blo, bhi,
                            0,1,2,3,4,5,6,7,8,9,10,11,12,13,14,15);
                acc[0][j] = __builtin_amdgcn_wmma_f32_16x16x32_f16(
                               false, af0, false, bf, (short)0, acc[0][j], false, false);
                acc[1][j] = __builtin_amdgcn_wmma_f32_16x16x32_f16(
                               false, af1, false, bf, (short)0, acc[1][j], false, false);
            }
        }

        // Fused epilogue. Tiles are 16-wide and 16-aligned -> whole tile in one
        // class. Reduce the 16 columns cross-lane, then one LDS atomic per row.
        #pragma unroll
        for (int j = 0; j < 4; ++j) {
            const int ncol = n0 + j * 16 + l15;
            const float aq = asq[ncol];
            const int cls  = (n0 + j * 16) >> 5;
            #pragma unroll
            for (int i = 0; i < 2; ++i) {
                #pragma unroll
                for (int r = 0; r < 8; ++r) {
                    float d2   = xs[i][r] + aq - 2.0f * acc[i][j][r];
                    float dist = __builtin_amdgcn_sqrtf(fmaxf(d2, 0.0f));
                    // sims = exp(-dist/t)  (in (0,1]); softmax numerator = exp(sims)
                    float sim  = __builtin_amdgcn_exp2f(-dist * invT * LOG2E);
                    float e    = __builtin_amdgcn_exp2f(sim * LOG2E);
                    #pragma unroll
                    for (int o = 1; o < 16; o <<= 1) e += __shfl_xor(e, o, 16);
                    if (l15 == 0)
                        atomicAdd(&clsAcc[mloc0 + i * 16 + r + (hi << 3)][cls], e);
                }
            }
        }
    }

    __syncthreads();
    // denominator = sum over all 2048 anchors = sum of the 64 class sums
    if (tid < M_TILE) {
        float s = 0.0f;
        #pragma unroll
        for (int c = 0; c < NUM_CLS; ++c) s += clsAcc[tid][c];
        denom[tid] = s;
    }
    __syncthreads();

    const int base_row = blockIdx.x * M_TILE;
    for (int i = tid; i < M_TILE * NUM_CLS; i += 256) {
        int row = i >> 6, c = i & 63;
        out[(size_t)(base_row + row) * NUM_CLS + c] = clsAcc[row][c] / denom[row];
    }
}

// ---------------------------------------------------------------------------
extern "C" void kernel_launch(void* const* d_in, const int* in_sizes, int n_in,
                              void* d_out, int out_size, void* d_ws, size_t ws_size,
                              hipStream_t stream)
{
    const float* emb  = (const float*)d_in[0];   // (8192, 1024) f32
    const float* anc  = (const float*)d_in[1];   // (64, 32, 1024) f32
    const int*   temp = (const int*)d_in[2];     // scalar temperature (=1)
    float*       out  = (float*)d_out;           // (8192, 64) f32

    char* ws = (char*)d_ws;
    _Float16* Xh  = (_Float16*)ws;                                       // 16 MB
    _Float16* Ah  = (_Float16*)(ws + (size_t)BATCH_N * EMBED_D * 2);     //  4 MB
    float*    xsq = (float*)(ws + (size_t)(BATCH_N + CA_N) * EMBED_D * 2);
    float*    asq = xsq + BATCH_N;

    cvt_rowsq_kernel<<<(BATCH_N * 32) / 256, 256, 0, stream>>>(emb, Xh, xsq, BATCH_N);
    cvt_rowsq_kernel<<<(CA_N * 32) / 256, 256, 0, stream>>>(anc, Ah, asq, CA_N);
    fused_anchor_kernel<<<BATCH_N / M_TILE, 256, 0, stream>>>(Xh, Ah, xsq, asq, temp, out);
}
